// LSTMOCR_77653008712090
// MI455X (gfx1250) — matrix operand order repaired and checked by reading, not compiled, not run on hardware
//
#include <hip/hip_runtime.h>

// ---------------------------------------------------------------------------
// MI455X (gfx1250, wave32) implementation of the CNN+LSTM OCR reference.
// bf16 activations/weights, f32 accumulation via v_wmma_f32_16x16x32_bf16.
// Each GEMM wave owns a 16x64 tile (4 accumulators): A fragment loaded once
// per K-chunk feeds 4 independent WMMAs (high XDL issue density, 1/4 the
// A-side VMEM traffic). K loops are compile-time and fully unrolled.
// ---------------------------------------------------------------------------

typedef __bf16 bf16;
typedef __attribute__((ext_vector_type(16))) __bf16 v16bf;
typedef __attribute__((ext_vector_type(8)))  __bf16 v8bf;
typedef __attribute__((ext_vector_type(8)))  float  v8f;

#define LEAK 0.01f
#define EPSB 1e-5f

static __device__ __forceinline__ float sigmoidf_(float x) {
  return 1.0f / (1.0f + __expf(-x));
}

static __device__ __forceinline__ v16bf load_a_frag(const bf16* ap) {
  v8bf lo = *(const v8bf*)(ap);
  v8bf hi = *(const v8bf*)(ap + 16);
  return __builtin_shufflevector(lo, hi, 0, 1, 2, 3, 4, 5, 6, 7, 8, 9, 10, 11, 12, 13, 14, 15);
}

// ---------------------------------------------------------------------------
// Pack a [K, Nsrc] fp32 weight matrix into bf16 WMMA B-fragments.
// Consumption layout: frag[((ntile*Kchunks + kc)*32 + lane)*16 + j]
//   with n = ntile*16 + (lane&15), k = kc*32 + (lane>>4)*8 + (j&7) + (j>>3)*16.
// ---------------------------------------------------------------------------
__global__ void pack_weights_kernel(const float* __restrict__ w, bf16* __restrict__ wpack,
                                    int K, int Nsrc, int Kchunks, int Ntiles) {
  int idx = blockIdx.x * blockDim.x + threadIdx.x;
  int total = Ntiles * Kchunks * 512;
  if (idx >= total) return;
  int j    = idx & 15;
  int lane = (idx >> 4) & 31;
  int kc   = (idx >> 9) % Kchunks;
  int nt   = (idx >> 9) / Kchunks;
  int n = nt * 16 + (lane & 15);
  int k = kc * 32 + (lane >> 4) * 8 + (j & 7) + (j >> 3) * 16;
  float v = (k < K && n < Nsrc) ? w[(size_t)k * Nsrc + n] : 0.0f;
  wpack[idx] = (bf16)v;
}

// ---------------------------------------------------------------------------
// conv0 (Cin=1, Cout=64) fused: conv3x3 SAME + BN + leaky + maxpool2x2.
// One block per pooled output pixel (b,yo,xo); 64 threads = channels.
// 4x4 input patch staged through LDS.
// ---------------------------------------------------------------------------
__global__ void conv0_kernel(const float* __restrict__ x, const float* __restrict__ w,
                             const float* __restrict__ cb, const float* __restrict__ g,
                             const float* __restrict__ bb, const float* __restrict__ mm,
                             const float* __restrict__ mv, bf16* __restrict__ out) {
  int blk = blockIdx.x;               // b*30*90 + yo*90 + xo
  int xo = blk % 90;
  int t1 = blk / 90;
  int yo = t1 % 30;
  int b  = t1 / 30;
  __shared__ float patch[4][4];
  int tid = threadIdx.x;
  if (tid < 16) {
    int dy = tid >> 2, dx = tid & 3;
    int yi = yo * 2 - 1 + dy, xi = xo * 2 - 1 + dx;
    float v = 0.0f;
    if (yi >= 0 && yi < 60 && xi >= 0 && xi < 180)
      v = x[(size_t)(b * 60 + yi) * 180 + xi];
    patch[dy][dx] = v;
  }
  __syncthreads();
  int c = tid;  // 0..63
  float wv[9];
#pragma unroll
  for (int r = 0; r < 9; ++r) wv[r] = w[r * 64 + c];
  float s  = g[c] * rsqrtf(mv[c] + EPSB);
  float sh = (cb[c] - mm[c]) * s + bb[c];
  float best = -3.0e38f;
#pragma unroll
  for (int py = 0; py < 2; ++py)
#pragma unroll
    for (int px = 0; px < 2; ++px) {
      float acc = 0.0f;
#pragma unroll
      for (int kh = 0; kh < 3; ++kh)
#pragma unroll
        for (int kw = 0; kw < 3; ++kw)
          acc += patch[py + kh][px + kw] * wv[kh * 3 + kw];
      float v = acc * s + sh;
      v = (v < 0.0f) ? LEAK * v : v;
      best = fmaxf(best, v);
    }
  out[(size_t)blk * 64 + c] = (bf16)best;
}

// ---------------------------------------------------------------------------
// Implicit-GEMM conv (3x3 SAME) + BN + leaky. One wave (32 threads) per block;
// wave computes a 16(M) x 64(N) tile: 4 accumulators, A loaded once per
// K-chunk, 4 independent WMMAs per chunk. K loop fully unrolled.
// grid = (Mtiles, COUT/64).
// ---------------------------------------------------------------------------
template <int KCHUNKS, int CIN, int COUT>
__global__ void conv_wmma_kernel(const bf16* __restrict__ in, const bf16* __restrict__ wpack,
                                 const float* __restrict__ cb, const float* __restrict__ g,
                                 const float* __restrict__ bb, const float* __restrict__ mm,
                                 const float* __restrict__ mv, bf16* __restrict__ out,
                                 int Hin, int Win) {
  const int lane = threadIdx.x & 31;
  const int h = lane >> 4;
  const int mrow = lane & 15;
  const int mtile = blockIdx.x;
  const int ngrp = blockIdx.y;          // group of 4 consecutive N-tiles
  const int nt0 = ngrp * 4;

  // Decompose this lane's A-row pixel once.
  const int hw = Hin * Win;
  int p = mtile * 16 + mrow;
  int b = p / hw;
  int rem = p - b * hw;
  int y = rem / Win;
  int x = rem - y * Win;

  v16bf zero16 = {};
  v8f acc0 = {}, acc1 = {}, acc2 = {}, acc3 = {};

#pragma unroll
  for (int kc = 0; kc < KCHUNKS; ++kc) {
    constexpr int chunks_per_tap = CIN / 32;
    const int r  = kc / chunks_per_tap;           // filter tap 0..8 (compile-time after unroll)
    const int c0 = (kc % chunks_per_tap) * 32;    // channel base
    const int dh = r / 3 - 1;
    const int dw = r % 3 - 1;
    const int yy = y + dh, xx = x + dw;

    v16bf a;
    if (yy >= 0 && yy < Hin && xx >= 0 && xx < Win) {
      a = load_a_frag(in + ((size_t)(b * Hin + yy) * Win + xx) * CIN + c0 + h * 8);
    } else {
      a = zero16;
    }
    const bf16* bp0 = wpack + (((size_t)(nt0 + 0) * KCHUNKS + kc) * 32 + lane) * 16;
    const bf16* bp1 = wpack + (((size_t)(nt0 + 1) * KCHUNKS + kc) * 32 + lane) * 16;
    const bf16* bp2 = wpack + (((size_t)(nt0 + 2) * KCHUNKS + kc) * 32 + lane) * 16;
    const bf16* bp3 = wpack + (((size_t)(nt0 + 3) * KCHUNKS + kc) * 32 + lane) * 16;
    if (kc + 1 < KCHUNKS)
      __builtin_prefetch((const void*)(bp0 + 512), 0, 3);   // global_prefetch_b8
    v16bf b0 = *(const v16bf*)bp0;
    v16bf b1 = *(const v16bf*)bp1;
    v16bf b2 = *(const v16bf*)bp2;
    v16bf b3 = *(const v16bf*)bp3;
    acc0 = __builtin_amdgcn_wmma_f32_16x16x32_bf16(false, a, false, b0, (short)0, acc0, false, false);
    acc1 = __builtin_amdgcn_wmma_f32_16x16x32_bf16(false, a, false, b1, (short)0, acc1, false, false);
    acc2 = __builtin_amdgcn_wmma_f32_16x16x32_bf16(false, a, false, b2, (short)0, acc2, false, false);
    acc3 = __builtin_amdgcn_wmma_f32_16x16x32_bf16(false, a, false, b3, (short)0, acc3, false, false);
  }

  // Fused inference BN + leaky ReLU; store bf16 NHWC.
  v8f accs[4] = {acc0, acc1, acc2, acc3};
#pragma unroll
  for (int t = 0; t < 4; ++t) {
    const int n = (nt0 + t) * 16 + mrow;
    float s  = g[n] * rsqrtf(mv[n] + EPSB);
    float sh = (cb[n] - mm[n]) * s + bb[n];
#pragma unroll
    for (int r8 = 0; r8 < 8; ++r8) {
      int pp = mtile * 16 + r8 + 8 * h;
      float v = accs[t][r8] * s + sh;
      v = (v < 0.0f) ? LEAK * v : v;
      out[(size_t)pp * COUT + n] = (bf16)v;
    }
  }
}

// ---------------------------------------------------------------------------
// 2x2 maxpool, stride 2, SAME (partial windows at the bottom/right edges).
// ---------------------------------------------------------------------------
__global__ void maxpool_kernel(const bf16* __restrict__ in, bf16* __restrict__ out,
                               int H, int W, int C, int Ho, int Wo, int total) {
  int idx = blockIdx.x * blockDim.x + threadIdx.x;
  if (idx >= total) return;
  int c = idx % C;
  int t = idx / C;
  int xo = t % Wo; t /= Wo;
  int yo = t % Ho;
  int b  = t / Ho;
  float best = -3.0e38f;
#pragma unroll
  for (int dy = 0; dy < 2; ++dy)
#pragma unroll
    for (int dx = 0; dx < 2; ++dx) {
      int yi = 2 * yo + dy, xi = 2 * xo + dx;
      if (yi < H && xi < W)
        best = fmaxf(best, (float)in[((size_t)(b * H + yi) * W + xi) * C + c]);
    }
  out[idx] = (bf16)best;
}

// pool3 [B,4,12,64] -> xseq [T=12, B, D=256], d = h*64 + c (transpose(0,2,1,3))
__global__ void build_xseq_kernel(const bf16* __restrict__ pool3, bf16* __restrict__ xseq) {
  int idx = blockIdx.x * blockDim.x + threadIdx.x;  // 12*256*256
  if (idx >= 12 * 256 * 256) return;
  int d = idx & 255;
  int b = (idx >> 8) & 255;
  int t = idx >> 16;
  int hh = d >> 6, c = d & 63;
  xseq[idx] = pool3[(((size_t)b * 4 + hh) * 12 + t) * 64 + c];
}

__global__ void zero_kernel(unsigned int* __restrict__ p, int n) {
  int i = blockIdx.x * blockDim.x + threadIdx.x;
  if (i < n) p[i] = 0u;
}

// ---------------------------------------------------------------------------
// LSTM gate GEMM: Z[M=256, N=512] = concat(A0, A1) @ Wpack + bias (fp32 out).
// A0 covers K-chunks [0, K0), A1 the rest. One wave per block, 16x64 tile.
// grid = (16 mtiles, 8 ngroups), block = 32.
// ---------------------------------------------------------------------------
template <int KCH, int K0>
__global__ void gate_wmma_kernel(const bf16* __restrict__ A0, int ldA0,
                                 const bf16* __restrict__ A1, int ldA1,
                                 const bf16* __restrict__ wpack, const float* __restrict__ bias,
                                 float* __restrict__ Z) {
  const int lane = threadIdx.x & 31;
  const int h = lane >> 4;
  const int mrow = lane & 15;
  const int mtile = blockIdx.x;
  const int nt0 = blockIdx.y * 4;
  const int row = mtile * 16 + mrow;

  v8f acc0 = {}, acc1 = {}, acc2 = {}, acc3 = {};
#pragma unroll
  for (int kc = 0; kc < KCH; ++kc) {
    const bf16* ap;
    if (kc < K0) ap = A0 + (size_t)row * ldA0 + kc * 32 + h * 8;
    else         ap = A1 + (size_t)row * ldA1 + (kc - K0) * 32 + h * 8;
    v16bf a = load_a_frag(ap);
    v16bf b0 = *(const v16bf*)(wpack + (((size_t)(nt0 + 0) * KCH + kc) * 32 + lane) * 16);
    v16bf b1 = *(const v16bf*)(wpack + (((size_t)(nt0 + 1) * KCH + kc) * 32 + lane) * 16);
    v16bf b2 = *(const v16bf*)(wpack + (((size_t)(nt0 + 2) * KCH + kc) * 32 + lane) * 16);
    v16bf b3 = *(const v16bf*)(wpack + (((size_t)(nt0 + 3) * KCH + kc) * 32 + lane) * 16);
    acc0 = __builtin_amdgcn_wmma_f32_16x16x32_bf16(false, a, false, b0, (short)0, acc0, false, false);
    acc1 = __builtin_amdgcn_wmma_f32_16x16x32_bf16(false, a, false, b1, (short)0, acc1, false, false);
    acc2 = __builtin_amdgcn_wmma_f32_16x16x32_bf16(false, a, false, b2, (short)0, acc2, false, false);
    acc3 = __builtin_amdgcn_wmma_f32_16x16x32_bf16(false, a, false, b3, (short)0, acc3, false, false);
  }
  v8f accs[4] = {acc0, acc1, acc2, acc3};
#pragma unroll
  for (int t = 0; t < 4; ++t) {
    const int n = (nt0 + t) * 16 + mrow;
    float bn = bias[n];
#pragma unroll
    for (int r8 = 0; r8 < 8; ++r8)
      Z[(size_t)(mtile * 16 + r8 + 8 * h) * 512 + n] = accs[t][r8] + bn;
  }
}

// TF LSTMCell pointwise: gates [i, j, f, o], forget_bias = 1.0.
__global__ void lstm_elem_kernel(const float* __restrict__ Z, float* __restrict__ c,
                                 bf16* __restrict__ hbf, bf16* __restrict__ hcopy) {
  int idx = blockIdx.x * blockDim.x + threadIdx.x;  // 256*128
  if (idx >= 256 * 128) return;
  int b = idx >> 7, j = idx & 127;
  const float* z = Z + (size_t)b * 512;
  float zi = z[j], zj = z[128 + j], zf = z[256 + j], zo = z[384 + j];
  float cn = c[idx] * sigmoidf_(zf + 1.0f) + sigmoidf_(zi) * tanhf(zj);
  float hv = tanhf(cn) * sigmoidf_(zo);
  c[idx] = cn;
  bf16 hb = (bf16)hv;
  hbf[idx] = hb;
  if (hcopy) hcopy[idx] = hb;
}

// ---------------------------------------------------------------------------
// logits = hseq[3072,128] @ Wout(pad 37->48) + bout, fp32 out [12,256,37].
// One wave per block handles all 3 N-tiles; K=4 chunks fully unrolled.
// ---------------------------------------------------------------------------
__global__ void out_wmma_kernel(const bf16* __restrict__ hseq, const bf16* __restrict__ wpack,
                                const float* __restrict__ bout, float* __restrict__ out) {
  const int lane = threadIdx.x & 31;
  const int h = lane >> 4;
  const int mrow = lane & 15;
  const int mtile = blockIdx.x;
  const int row = mtile * 16 + mrow;

  v8f acc0 = {}, acc1 = {}, acc2 = {};
#pragma unroll
  for (int kc = 0; kc < 4; ++kc) {
    v16bf a = load_a_frag(hseq + (size_t)row * 128 + kc * 32 + h * 8);
    v16bf b0 = *(const v16bf*)(wpack + (((size_t)0 * 4 + kc) * 32 + lane) * 16);
    v16bf b1 = *(const v16bf*)(wpack + (((size_t)1 * 4 + kc) * 32 + lane) * 16);
    v16bf b2 = *(const v16bf*)(wpack + (((size_t)2 * 4 + kc) * 32 + lane) * 16);
    acc0 = __builtin_amdgcn_wmma_f32_16x16x32_bf16(false, a, false, b0, (short)0, acc0, false, false);
    acc1 = __builtin_amdgcn_wmma_f32_16x16x32_bf16(false, a, false, b1, (short)0, acc1, false, false);
    acc2 = __builtin_amdgcn_wmma_f32_16x16x32_bf16(false, a, false, b2, (short)0, acc2, false, false);
  }
  v8f accs[3] = {acc0, acc1, acc2};
#pragma unroll
  for (int t = 0; t < 3; ++t) {
    const int n = t * 16 + mrow;
    if (n < 37) {
      float bn = bout[n];
#pragma unroll
      for (int r8 = 0; r8 < 8; ++r8)
        out[(size_t)(mtile * 16 + r8 + 8 * h) * 37 + n] = accs[t][r8] + bn;
    }
  }
}

// ---------------------------------------------------------------------------
extern "C" void kernel_launch(void* const* d_in, const int* in_sizes, int n_in,
                              void* d_out, int out_size, void* d_ws, size_t ws_size,
                              hipStream_t stream) {
  (void)in_sizes; (void)n_in; (void)out_size; (void)ws_size;

  const float* x = (const float*)d_in[0];
  const float *w[4], *cb[4], *g[4], *bb[4], *mm[4], *mv[4];
  for (int i = 0; i < 4; ++i) {
    w[i]  = (const float*)d_in[1 + 6 * i];
    cb[i] = (const float*)d_in[2 + 6 * i];
    g[i]  = (const float*)d_in[3 + 6 * i];
    bb[i] = (const float*)d_in[4 + 6 * i];
    mm[i] = (const float*)d_in[5 + 6 * i];
    mv[i] = (const float*)d_in[6 + 6 * i];
  }
  const float* lW0  = (const float*)d_in[25];
  const float* lb0  = (const float*)d_in[26];
  const float* lW1  = (const float*)d_in[27];
  const float* lb1  = (const float*)d_in[28];
  const float* Wout = (const float*)d_in[29];
  const float* bout = (const float*)d_in[30];
  float* out = (float*)d_out;

  // --- workspace bump allocator
  char* base = (char*)d_ws;
  size_t off = 0;
  auto alloc = [&](size_t bytes) -> char* {
    off = (off + 255) & ~(size_t)255;
    char* p = base + off;
    off += bytes;
    return p;
  };

  bf16* wp1  = (bf16*)alloc((size_t)8  * 18 * 512 * 2);   // conv1 K=576,  N=128
  bf16* wp2  = (bf16*)alloc((size_t)8  * 36 * 512 * 2);   // conv2 K=1152, N=128
  bf16* wp3  = (bf16*)alloc((size_t)4  * 36 * 512 * 2);   // conv3 K=1152, N=64
  bf16* wpl0 = (bf16*)alloc((size_t)32 * 12 * 512 * 2);   // lW0 K=384, N=512
  bf16* wpl1 = (bf16*)alloc((size_t)32 * 8  * 512 * 2);   // lW1 K=256, N=512
  bf16* wpo  = (bf16*)alloc((size_t)3  * 4  * 512 * 2);   // Wout K=128, N=37->48
  float* Z0  = (float*)alloc((size_t)256 * 512 * 4);
  float* Z1  = (float*)alloc((size_t)256 * 512 * 4);
  float* c0  = (float*)alloc((size_t)256 * 128 * 4);      // state arena: c0,c1,h0,h1
  float* c1  = (float*)alloc((size_t)256 * 128 * 4);      //   contiguous (each 256B-mult)
  bf16* h0   = (bf16*)alloc((size_t)256 * 128 * 2);
  bf16* h1   = (bf16*)alloc((size_t)256 * 128 * 2);
  bf16* hseq = (bf16*)alloc((size_t)12 * 256 * 128 * 2);
  bf16* xseq = (bf16*)alloc((size_t)12 * 256 * 256 * 2);
  bf16* pool3 = (bf16*)alloc((size_t)256 * 4 * 12 * 64 * 2);
  bf16* bigA = (bf16*)alloc((size_t)256 * 30 * 90 * 64 * 2);    // act1 / conv2out / conv3out
  bf16* bigB = (bf16*)alloc((size_t)256 * 30 * 90 * 128 * 2);   // conv1out / pool2
  bf16* bigC = (bf16*)alloc((size_t)256 * 15 * 45 * 128 * 2);   // pool1

  auto cdiv = [](int a, int b) { return (a + b - 1) / b; };

  // --- pack all GEMM weights into WMMA B-fragment order (bf16)
  pack_weights_kernel<<<cdiv(8 * 18 * 512, 256), 256, 0, stream>>>(w[1], wp1, 576, 128, 18, 8);
  pack_weights_kernel<<<cdiv(8 * 36 * 512, 256), 256, 0, stream>>>(w[2], wp2, 1152, 128, 36, 8);
  pack_weights_kernel<<<cdiv(4 * 36 * 512, 256), 256, 0, stream>>>(w[3], wp3, 1152, 64, 36, 4);
  pack_weights_kernel<<<cdiv(32 * 12 * 512, 256), 256, 0, stream>>>(lW0, wpl0, 384, 512, 12, 32);
  pack_weights_kernel<<<cdiv(32 * 8 * 512, 256), 256, 0, stream>>>(lW1, wpl1, 256, 512, 8, 32);
  pack_weights_kernel<<<cdiv(3 * 4 * 512, 256), 256, 0, stream>>>(Wout, wpo, 128, 37, 4, 3);

  // --- conv0 fused (conv+BN+leaky+pool) -> bigA [256,30,90,64] bf16
  conv0_kernel<<<256 * 30 * 90, 64, 0, stream>>>(x, w[0], cb[0], g[0], bb[0], mm[0], mv[0], bigA);

  // --- conv1 (WMMA) -> bigB [256,30,90,128]; pool -> bigC [256,15,45,128]
  conv_wmma_kernel<18, 64, 128><<<dim3(43200, 2), 32, 0, stream>>>(
      bigA, wp1, cb[1], g[1], bb[1], mm[1], mv[1], bigB, 30, 90);
  maxpool_kernel<<<cdiv(256 * 15 * 45 * 128, 256), 256, 0, stream>>>(bigB, bigC, 30, 90, 128, 15, 45,
                                                                     256 * 15 * 45 * 128);
  // --- conv2 (WMMA) -> bigA [256,15,45,128]; pool -> bigB [256,8,23,128]
  conv_wmma_kernel<36, 128, 128><<<dim3(10800, 2), 32, 0, stream>>>(
      bigC, wp2, cb[2], g[2], bb[2], mm[2], mv[2], bigA, 15, 45);
  maxpool_kernel<<<cdiv(256 * 8 * 23 * 128, 256), 256, 0, stream>>>(bigA, bigB, 15, 45, 128, 8, 23,
                                                                    256 * 8 * 23 * 128);
  // --- conv3 (WMMA) -> bigA [256,8,23,64]; pool -> pool3 [256,4,12,64]
  conv_wmma_kernel<36, 128, 64><<<dim3(2944, 1), 32, 0, stream>>>(
      bigB, wp3, cb[3], g[3], bb[3], mm[3], mv[3], bigA, 8, 23);
  maxpool_kernel<<<cdiv(256 * 4 * 12 * 64, 256), 256, 0, stream>>>(bigA, pool3, 8, 23, 64, 4, 12,
                                                                   256 * 4 * 12 * 64);

  // --- features -> time-major sequence [12,256,256]
  build_xseq_kernel<<<cdiv(12 * 256 * 256, 256), 256, 0, stream>>>(pool3, xseq);

  // --- zero LSTM state (c0,c1 fp32 + h0,h1 bf16 are contiguous: 393216 B)
  zero_kernel<<<cdiv(98304, 256), 256, 0, stream>>>((unsigned int*)c0, 98304);

  // --- 12 LSTM steps, 2 layers each (all WMMA gate GEMMs)
  for (int t = 0; t < 12; ++t) {
    gate_wmma_kernel<12, 8><<<dim3(16, 8), 32, 0, stream>>>(
        xseq + (size_t)t * 256 * 256, 256, h0, 128, wpl0, lb0, Z0);
    lstm_elem_kernel<<<128, 256, 0, stream>>>(Z0, c0, h0, (bf16*)nullptr);
    gate_wmma_kernel<8, 4><<<dim3(16, 8), 32, 0, stream>>>(
        h0, 128, h1, 128, wpl1, lb1, Z1);
    lstm_elem_kernel<<<128, 256, 0, stream>>>(Z1, c1, h1, hseq + (size_t)t * 256 * 128);
  }

  // --- logits [12,256,37] fp32
  out_wmma_kernel<<<192, 32, 0, stream>>>(hseq, wpo, bout, out);
}
